// VQ_6193342841500
// MI455X (gfx1250) — matrix operand order repaired and checked by reading, not compiled, not run on hardware
//
#include <hip/hip_runtime.h>

// ---- VQ nearest-codebook-row kernel for gfx1250 (wave32, WMMA f16) ----
// x:   [N=262144, D=64] fp32
// emb: [D=64, K=512]    fp32 (emb[d*K + n])
// out: [N, D]           fp32 = codebook row of argmin_n ||x_m - e_n||^2
//
// dist(m,n) = ||x_m||^2 + ||e_n||^2 - 2 x_m.e_n ; the ||x_m||^2 term is
// row-constant, so argmin over n needs only (||e_n||^2 - 2 x_m.e_n).
// The dot products are a 262144x512x64 GEMM -> v_wmma_f32_16x16x32_f16,
// software-pipelined against ds_load_b128 B-tile fetches.

typedef __attribute__((ext_vector_type(16))) _Float16 v16h;
typedef __attribute__((ext_vector_type(8)))  _Float16 v8h;
typedef __attribute__((ext_vector_type(8)))  float    v8f;

#define VQ_D 64
#define VQ_K 512
#define NTILE (VQ_K / 16)   // 32 codebook column tiles
#define LROW 72             // f16 per LDS codebook row: 64 data + 8 pad (144B, 16B-aligned, bank-skewed)

// LDS layout (dynamic):
//   [0,     73728) : f16 codebook, transposed: ebf[n*LROW + d]
//   [73728, 75776) : float e2[512]  (||e_n||^2)
//   [75776, 76288) : int rowidx[8][16] (per-wave winning indices)
#define EBF_BYTES   (VQ_K * LROW * 2)
#define E2_OFF      EBF_BYTES
#define RIDX_OFF    (EBF_BYTES + VQ_K * 4)
#define SMEM_BYTES  (RIDX_OFF + 8 * 16 * 4)

__global__ __launch_bounds__(256)
void vq_wmma_kernel(const float* __restrict__ x,
                    const float* __restrict__ emb,
                    float* __restrict__ out)
{
    extern __shared__ char smem[];
    _Float16* ebf    = (_Float16*)smem;
    float*    e2     = (float*)(smem + E2_OFF);
    int*      rowidx = (int*)(smem + RIDX_OFF);

    const int tid = threadIdx.x;

    // ---- stage codebook into LDS: f16, transposed [n][d] ----
    for (int i = tid; i < VQ_D * VQ_K; i += 256) {
        int d = i >> 9;            // i / 512
        int n = i & (VQ_K - 1);    // i % 512
        ebf[n * LROW + d] = (_Float16)emb[i];
    }
    // ---- codebook squared norms (fp32, exact) ----
    for (int n = tid; n < VQ_K; n += 256) {
        float s = 0.f;
        #pragma unroll 8
        for (int d = 0; d < VQ_D; ++d) {
            float v = emb[d * VQ_K + n];
            s = fmaf(v, v, s);
        }
        e2[n] = s;
    }
    __syncthreads();

    const int wave = tid >> 5;     // 0..7
    const int lane = tid & 31;
    const int lo   = lane & 15;    // column / M-row selector per ISA layout
    const int hi   = lane >> 4;    // half-wave selector

    const long m0 = (long)blockIdx.x * 128 + (long)wave * 16;

    // ---- A fragments: 16x32 f16, K-chunks [0,32) and [32,64) ----
    // A layout (matches B): lanes 0-15 M=lane; VGPR v in 0..7 -> K = hi*16 + 2v{,+1}.
    // (v16h elem e -> VGPR e/2, half e&1)
    const float* xrow = x + (m0 + lo) * VQ_D;
    v16h afrag0, afrag1;
    #pragma unroll
    for (int half = 0; half < 2; ++half) {
        const int kb0 = hi * 16 + half * 8;
        #pragma unroll
        for (int j = 0; j < 8; ++j) {
            afrag0[half * 8 + j] = (_Float16)xrow[kb0 + j];
            afrag1[half * 8 + j] = (_Float16)xrow[32 + kb0 + j];
        }
    }

    float best[8];
    int   bidx[8];
    #pragma unroll
    for (int r = 0; r < 8; ++r) { best[r] = 3.4e38f; bidx[r] = 0; }

    // B 32x16 f16 layout: lane holds column n = nt*16+lo; VGPR v -> K = hi*16 + 2v{,+1}.
    // In the transposed LDS image those 16 halves are contiguous -> two b128 loads/fragment.
    auto loadB = [&](int nt, v16h& B0, v16h& B1, float& EN) {
        const int n = nt * 16 + lo;
        const v8h* p0 = (const v8h*)(ebf + n * LROW + hi * 16);        // d = hi*16 + 0..15
        const v8h* p1 = (const v8h*)(ebf + n * LROW + 32 + hi * 16);   // d = 32 + hi*16 + 0..15
        v8h l0 = p0[0], h0 = p0[1];
        v8h l1 = p1[0], h1 = p1[1];
        B0 = __builtin_shufflevector(l0, h0, 0,1,2,3,4,5,6,7,8,9,10,11,12,13,14,15);
        B1 = __builtin_shufflevector(l1, h1, 0,1,2,3,4,5,6,7,8,9,10,11,12,13,14,15);
        EN = e2[n & (VQ_K - 1)];   // masked so the tail prefetch stays in e2[]
    };

    auto compute = [&](int nt, v16h B0, v16h B1, float EN) {
        v8f c = {};
        c = __builtin_amdgcn_wmma_f32_16x16x32_f16(false, afrag0, false, B0,
                                                   (short)0, c, false, false);
        c = __builtin_amdgcn_wmma_f32_16x16x32_f16(false, afrag1, false, B1,
                                                   (short)0, c, false, false);
        const int n = nt * 16 + lo;
        #pragma unroll
        for (int r = 0; r < 8; ++r) {
            float dd = fmaf(-2.0f, c[r], EN);     // ||e_n||^2 - 2 x.e_n
            if (dd < best[r]) { best[r] = dd; bidx[r] = n; }  // n increasing -> first-min kept
        }
    };

    // ---- software-pipelined sweep: ping-pong B buffers, loads one tile ahead ----
    v16h b0a, b1a, b0b, b1b;
    float ena, enb;
    loadB(0, b0a, b1a, ena);
    #pragma unroll 2
    for (int nt = 0; nt < NTILE; nt += 2) {
        loadB(nt + 1, b0b, b1b, enb);     // prefetch odd tile
        compute(nt, b0a, b1a, ena);       // consume even tile
        loadB(nt + 2, b0a, b1a, ena);     // prefetch next even tile (tail read is in-bounds of smem, unused)
        compute(nt + 1, b0b, b1b, enb);   // consume odd tile
    }

    // ---- cross-lane argmin over the 16 columns in each half-wave ----
    #pragma unroll
    for (int off = 8; off > 0; off >>= 1) {
        #pragma unroll
        for (int r = 0; r < 8; ++r) {
            float od = __shfl_xor(best[r], off, 16);
            int   oi = __shfl_xor(bidx[r], off, 16);
            if (od < best[r] || (od == best[r] && oi < bidx[r])) {
                best[r] = od; bidx[r] = oi;
            }
        }
    }
    if (lo == 0) {
        #pragma unroll
        for (int r = 0; r < 8; ++r)
            rowidx[wave * 16 + hi * 8 + r] = bidx[r];   // row M = hi*8 + r (C layout)
    }

    // ---- gather fp32 codebook rows and write output (coalesced stores) ----
    #pragma unroll 4
    for (int j = 0; j < 16; ++j) {
        const int idx = rowidx[wave * 16 + j];
        const float v0 = emb[lane * VQ_K + idx];          // d = lane       (L2-resident table)
        const float v1 = emb[(lane + 32) * VQ_K + idx];   // d = lane + 32
        float* orow = out + (m0 + j) * VQ_D;
        orow[lane]      = v0;
        orow[lane + 32] = v1;
    }
}

extern "C" void kernel_launch(void* const* d_in, const int* in_sizes, int n_in,
                              void* d_out, int out_size, void* d_ws, size_t ws_size,
                              hipStream_t stream) {
    const float* x   = (const float*)d_in[0];   // (64,64,64,64) fp32 = 262144 x 64
    const float* emb = (const float*)d_in[1];   // (64,512) fp32
    float* out = (float*)d_out;

    const int nrows  = in_sizes[0] / VQ_D;      // 262144
    const int blocks = nrows / 128;             // 128 rows per block (8 waves x 16)

    vq_wmma_kernel<<<blocks, 256, SMEM_BYTES, stream>>>(x, emb, out);
}
